// CNNBlock_52312701665417
// MI455X (gfx1250) — matrix operand order repaired and checked by reading, not compiled, not run on hardware
//
#include <hip/hip_runtime.h>
#include <stdint.h>

// ---------------- problem constants ----------------
#define BATCH 128
#define NSTAB 624
#define HID   256
#define CDIM  128
#define G     26
#define GG    676            // 26*26
#define ROWS  (BATCH*NSTAB)  // 79872
#define MTILES (ROWS/16)     // 4992
#define BN_COUNT (BATCH*GG)  // 86528
#define PT64  11             // ceil(676/64) pixel super-tiles per image

typedef __bf16 v16bf __attribute__((ext_vector_type(16)));
typedef float  v8f   __attribute__((ext_vector_type(8)));

struct BF16Frag { uint4 lo, hi; };

static __device__ __forceinline__ v16bf frag_cast(uint4 lo, uint4 hi) {
    BF16Frag f; f.lo = lo; f.hi = hi;
    return __builtin_bit_cast(v16bf, f);
}

static __device__ __forceinline__ unsigned short f2bf_bits(float f) {
    uint32_t u = __builtin_bit_cast(uint32_t, f);
    uint32_t r = u + 0x7FFFu + ((u >> 16) & 1u);   // round-to-nearest-even
    return (unsigned short)(r >> 16);
}

static __device__ __forceinline__ __bf16 f2bf(float f) {
    return __builtin_bit_cast(__bf16, f2bf_bits(f));
}

#define WMMA_BF16(A, B, C) \
    __builtin_amdgcn_wmma_f32_16x16x32_bf16(false, (A), false, (B), (short)0, (C), false, false)

// ---- CDNA5 async global->LDS copy (tracked by ASYNCcnt) ----
typedef __attribute__((address_space(3))) unsigned char lds_u8_t;

static __device__ __forceinline__ uint32_t lds_addr32(const void* p) {
    return (uint32_t)(uintptr_t)(lds_u8_t*)(p);   // raw 32-bit LDS offset
}

static __device__ __forceinline__ void async_copy_b128(uint32_t lds_off, const void* gptr) {
    unsigned long long ga = (unsigned long long)(uintptr_t)gptr;
    asm volatile("global_load_async_to_lds_b128 %0, %1, off"
                 :: "v"(lds_off), "v"(ga) : "memory");
}

static __device__ __forceinline__ void wait_asynccnt0() {
    asm volatile("s_wait_asynccnt 0x0" ::: "memory");
}

// load a 16x32 bf16 A/B-fragment half-pair (two 16B chunks) from a generic pointer
static __device__ __forceinline__ v16bf frag_ld_a(const __bf16* base) {
    uint4 lo = *(const uint4*)(base);
    uint4 hi = *(const uint4*)(base + 8);
    return frag_cast(lo, hi);
}
static __device__ __forceinline__ v16bf frag_ld_b(const __bf16* base) {
    const uint4* p = (const uint4*)base;
    return frag_cast(p[0], p[1]);
}

// ---------------- x: fp32 -> bf16 (one pass, memory bound) ----------------
__global__ __launch_bounds__(256) void convert_x_kernel(
    const float* __restrict__ x, __bf16* __restrict__ xbf)
{
    size_t i8 = ((size_t)blockIdx.x * 256 + threadIdx.x) * 8;   // 8 elems/thread
    unsigned short h[8];
#pragma unroll
    for (int i = 0; i < 8; ++i) h[i] = f2bf_bits(x[i8 + i]);
    uint4 r;
    r.x = (uint32_t)h[0] | ((uint32_t)h[1] << 16);
    r.y = (uint32_t)h[2] | ((uint32_t)h[3] << 16);
    r.z = (uint32_t)h[4] | ((uint32_t)h[5] << 16);
    r.w = (uint32_t)h[6] | ((uint32_t)h[7] << 16);
    *(uint4*)(xbf + i8) = r;
}

// ---------------- weight prep: fp32 -> bf16, transposed layouts ----------------
__global__ __launch_bounds__(256) void prep_weights_kernel(
    const float* __restrict__ Win, const float* __restrict__ w0,
    const float* __restrict__ w1, const float* __restrict__ Wout,
    __bf16* __restrict__ WinT, __bf16* __restrict__ w0r,
    __bf16* __restrict__ w1r, __bf16* __restrict__ WoutT)
{
    int i = blockIdx.x * 256 + threadIdx.x;
    if (i < CDIM * HID) {                 // WinT[n][k], n<128,k<256
        int n = i / HID, k = i - n * HID;
        WinT[i] = f2bf(Win[k * CDIM + n]);
    }
    if (i < 9 * CDIM * CDIM) {            // w{0,1}r[tap][o][i]
        int tap = i / (CDIM * CDIM);
        int rem = i - tap * (CDIM * CDIM);
        int o = rem / CDIM, ii = rem - o * CDIM;
        int src = (o * CDIM + ii) * 9 + tap;   // OIHW flat
        w0r[i] = f2bf(w0[src]);
        w1r[i] = f2bf(w1[src]);
    }
    if (i < HID * CDIM) {                 // WoutT[n][k], n<256,k<128
        int n = i / CDIM, k = i - n * CDIM;
        WoutT[i] = f2bf(Wout[k * HID + n]);
    }
}

// ---------------- proj_in: h = x @ W_in + b_in, scattered into grid ----------------
// block = 64 rows x 128 cols; wave (mg,ng) -> 2x2 register tile of 16x16 WMMA tiles
__global__ __launch_bounds__(256) void proj_in_kernel(
    const __bf16* __restrict__ xbf, const __bf16* __restrict__ WinT,
    const float* __restrict__ b_in,
    float* __restrict__ gridA, __bf16* __restrict__ gbf)
{
    const int rowbase = blockIdx.x * 64;
    const int lane = threadIdx.x & 31;
    const int wid  = threadIdx.x >> 5;
    const int hs   = lane >> 4;
    const int lm   = lane & 15;
    const int mg   = wid >> 2;           // 0,1: M-pair
    const int ng   = wid & 3;            // 0..3: N-pair

    const int gr0 = rowbase + (2 * mg) * 16 + lm;
    const __bf16* xr0 = xbf + (size_t)gr0 * HID;
    const __bf16* xr1 = xr0 + (size_t)16 * HID;
    const int n0 = (2 * ng) * 16 + lm;
    const int n1 = n0 + 16;
    const __bf16* br0 = WinT + (size_t)n0 * HID;
    const __bf16* br1 = WinT + (size_t)n1 * HID;

    v8f acc[2][2] = {};
#pragma unroll
    for (int ks = 0; ks < 8; ++ks) {     // K = 256, step 32
        const int kb = ks * 32;
        v16bf a0 = frag_ld_a(xr0 + kb + 16 * hs);
        v16bf a1 = frag_ld_a(xr1 + kb + 16 * hs);
        v16bf b0 = frag_ld_b(br0 + kb + 16 * hs);
        v16bf b1 = frag_ld_b(br1 + kb + 16 * hs);
        acc[0][0] = WMMA_BF16(a0, b0, acc[0][0]);
        acc[0][1] = WMMA_BF16(a0, b1, acc[0][1]);
        acc[1][0] = WMMA_BF16(a1, b0, acc[1][0]);
        acc[1][1] = WMMA_BF16(a1, b1, acc[1][1]);
    }

    const float bias0 = b_in[n0];
    const float bias1 = b_in[n1];
#pragma unroll
    for (int mt = 0; mt < 2; ++mt) {
#pragma unroll
        for (int r = 0; r < 8; ++r) {
            int gr = rowbase + (2 * mg + mt) * 16 + r + 8 * hs;
            int bb = gr / NSTAB;
            int ss = gr - bb * NSTAB;    // cell index == stabilizer index
            size_t o = ((size_t)bb * GG + ss) * CDIM;
            float v0 = acc[mt][0][r] + bias0;
            float v1 = acc[mt][1][r] + bias1;
            gridA[o + n0] = v0;  gbf[o + n0] = f2bf(v0);
            gridA[o + n1] = v1;  gbf[o + n1] = f2bf(v1);
        }
    }
}

// ---------------- dilated 3x3 conv as implicit GEMM + BN stats ----------------
// block = 64 pixels x 128 out-ch; halo staged in LDS via async loads; 2x2 wave tiling
#define SROWS 8
#define ROWBYTES (G*CDIM*2)     // 6656 bytes per staged grid row
#define ROWVECS  (ROWBYTES/16)  // 416 uint4 per row

__global__ __launch_bounds__(256) void conv_kernel(
    const __bf16* __restrict__ gbf, const __bf16* __restrict__ wr,
    const float* __restrict__ cbias,
    float* __restrict__ y, float* __restrict__ stats, int dil)
{
    __shared__ __align__(16) unsigned char sA[SROWS * ROWBYTES];   // 53248 B
    __shared__ float s_sum[CDIM];
    __shared__ float s_sq[CDIM];

    const int b    = blockIdx.x / PT64;
    const int pt   = blockIdx.x - b * PT64;
    const int lane = threadIdx.x & 31;
    const int wid  = threadIdx.x >> 5;
    const int hs   = lane >> 4;
    const int lm   = lane & 15;
    const int mg   = wid >> 2;            // 0,1: pixel-tile pair
    const int ng   = wid & 3;             // 0..3: out-channel pair

    if (threadIdx.x < CDIM) { s_sum[threadIdx.x] = 0.f; s_sq[threadIdx.x] = 0.f; }

    // ---- stage halo rows [y0-d, y1+d] into LDS (zeros for off-grid rows) ----
    const int p0 = pt * 64;
    const int y0 = p0 / G;
    int y1 = (p0 + 63) / G; if (y1 > G - 1) y1 = G - 1;
    const int ylo = y0 - dil;                 // unclamped; slot = row - ylo
    const int nslots = (y1 + dil) - ylo + 1;  // <= 8
    const int total = nslots * ROWVECS;
    const uint4 z4 = make_uint4(0u, 0u, 0u, 0u);
    for (int t = threadIdx.x; t < total; t += 256) {
        int slot = t / ROWVECS;
        int rem  = t - slot * ROWVECS;
        int row  = ylo + slot;
        uint32_t loff = (uint32_t)(slot * ROWBYTES + rem * 16);
        if (row >= 0 && row < G) {
            const unsigned char* gsrc =
                (const unsigned char*)(gbf + ((size_t)b * GG + (size_t)row * G) * CDIM) + rem * 16;
            async_copy_b128(lds_addr32(sA + loff), gsrc);
        } else {
            *(uint4*)(sA + loff) = z4;        // zero pad off-grid rows
        }
    }
    wait_asynccnt0();
    __syncthreads();

    const int pA0 = p0 + (2 * mg) * 16 + lm;  // A rows for the two M fragments
    const int pA1 = pA0 + 16;
    const int py0 = pA0 / G, px0 = pA0 - py0 * G;
    const int py1 = pA1 / G, px1 = pA1 - py1 * G;
    const int n0  = (2 * ng) * 16 + lm;       // the two out-channel columns
    const int n1  = n0 + 16;
    const __bf16* wb0 = wr + (size_t)n0 * CDIM;
    const __bf16* wb1 = wr + (size_t)n1 * CDIM;

    v8f acc[2][2] = {};
#pragma unroll
    for (int tap = 0; tap < 9; ++tap) {
        const int dy = (tap / 3 - 1) * dil;
        const int dx = (tap % 3 - 1) * dil;
        const int iy0 = py0 + dy, ix0 = px0 + dx;
        const int iy1 = py1 + dy, ix1 = px1 + dx;
        const bool v0 = (pA0 < GG) && (ix0 >= 0) && (ix0 < G) && (iy0 >= 0) && (iy0 < G);
        const bool v1 = (pA1 < GG) && (ix1 >= 0) && (ix1 < G) && (iy1 >= 0) && (iy1 < G);
        const __bf16* wt0 = wb0 + (size_t)tap * CDIM * CDIM;
        const __bf16* wt1 = wb1 + (size_t)tap * CDIM * CDIM;
#pragma unroll
        for (int kc = 0; kc < 4; ++kc) {      // 128 in-channels, step 32
            const int kb = kc * 32;
            uint4 a0lo = z4, a0hi = z4, a1lo = z4, a1hi = z4;
            if (v0) {
                const unsigned char* ab =
                    sA + (size_t)(iy0 - ylo) * ROWBYTES + (size_t)ix0 * (CDIM * 2) + (kb + 16 * hs) * 2;
                a0lo = *(const uint4*)(ab);          // ds_load_b128
                a0hi = *(const uint4*)(ab + 16);
            }
            if (v1) {
                const unsigned char* ab =
                    sA + (size_t)(iy1 - ylo) * ROWBYTES + (size_t)ix1 * (CDIM * 2) + (kb + 16 * hs) * 2;
                a1lo = *(const uint4*)(ab);
                a1hi = *(const uint4*)(ab + 16);
            }
            v16bf a0 = frag_cast(a0lo, a0hi);
            v16bf a1 = frag_cast(a1lo, a1hi);
            v16bf b0 = frag_ld_b(wt0 + kb + 16 * hs);
            v16bf b1 = frag_ld_b(wt1 + kb + 16 * hs);
            acc[0][0] = WMMA_BF16(a0, b0, acc[0][0]);
            acc[0][1] = WMMA_BF16(a0, b1, acc[0][1]);
            acc[1][0] = WMMA_BF16(a1, b0, acc[1][0]);
            acc[1][1] = WMMA_BF16(a1, b1, acc[1][1]);
        }
    }

    const float bias0 = cbias[n0];
    const float bias1 = cbias[n1];
    float ls0 = 0.f, lq0 = 0.f, ls1 = 0.f, lq1 = 0.f;
#pragma unroll
    for (int mt = 0; mt < 2; ++mt) {
#pragma unroll
        for (int r = 0; r < 8; ++r) {
            int p = p0 + (2 * mg + mt) * 16 + r + 8 * hs;
            if (p < GG) {
                size_t o = ((size_t)b * GG + p) * CDIM;
                float w0v = acc[mt][0][r] + bias0;
                float w1v = acc[mt][1][r] + bias1;
                y[o + n0] = w0v;  ls0 += w0v;  lq0 += w0v * w0v;
                y[o + n1] = w1v;  ls1 += w1v;  lq1 += w1v * w1v;
            }
        }
    }
    atomicAdd(&s_sum[n0], ls0);               // ds_add_f32
    atomicAdd(&s_sq[n0],  lq0);
    atomicAdd(&s_sum[n1], ls1);
    atomicAdd(&s_sq[n1],  lq1);
    __syncthreads();
    if (threadIdx.x < CDIM) {
        atomicAdd(&stats[threadIdx.x],        s_sum[threadIdx.x]);
        atomicAdd(&stats[CDIM + threadIdx.x], s_sq[threadIdx.x]);
    }
}

// ---------------- BN(train) + ReLU + residual; refresh fp32 + bf16 grids ----------------
__global__ __launch_bounds__(256) void bn_apply_kernel(
    const float* __restrict__ y, const float* __restrict__ stats,
    const float* __restrict__ g, const float* __restrict__ beta,
    float* __restrict__ gridA, __bf16* __restrict__ gbf)
{
    size_t idx = (size_t)blockIdx.x * 256 + threadIdx.x;   // 11075584 elements exactly
    int ch = (int)(idx & (CDIM - 1));
    const float inv_cnt = 1.0f / (float)BN_COUNT;
    float mu  = stats[ch] * inv_cnt;
    float var = stats[CDIM + ch] * inv_cnt - mu * mu;
    float inv = __frsqrt_rn(var + 1e-5f);
    float yn  = (y[idx] - mu) * inv * g[ch] + beta[ch];
    float v   = gridA[idx] + fmaxf(yn, 0.f);
    gridA[idx] = v;
    gbf[idx]   = f2bf(v);
}

// ---------------- gather + proj_out WMMA + residual + LayerNorm ----------------
__global__ __launch_bounds__(256) void proj_out_ln_kernel(
    const float* __restrict__ x, const __bf16* __restrict__ gbf,
    const __bf16* __restrict__ WoutT, const float* __restrict__ b_out,
    const float* __restrict__ ln_g, const float* __restrict__ ln_b,
    float* __restrict__ out)
{
    __shared__ float zsh[16][HID];
    __shared__ float red1[16][17];
    __shared__ float red2[16][17];

    const int tile = blockIdx.x;
    const int lane = threadIdx.x & 31;
    const int wid  = threadIdx.x >> 5;
    const int hs   = lane >> 4;
    const int lm   = lane & 15;

    const int grA = tile * 16 + lm;
    const int bb  = grA / NSTAB;
    const int ss  = grA - bb * NSTAB;
    const __bf16* arow = gbf + ((size_t)bb * GG + ss) * CDIM;

    const int n0 = wid * 16 + lm;        // wave covers N tiles wid and wid+8
    const int n1 = n0 + 128;
    const __bf16* brow0 = WoutT + (size_t)n0 * CDIM;
    const __bf16* brow1 = WoutT + (size_t)n1 * CDIM;

    v8f acc0 = {}, acc1 = {};
#pragma unroll
    for (int ks = 0; ks < 4; ++ks) {     // K = 128, step 32
        const int kb = ks * 32;
        v16bf a = frag_ld_a(arow + kb + 16 * hs);
        v16bf b0 = frag_ld_b(brow0 + kb + 16 * hs);
        acc0 = WMMA_BF16(a, b0, acc0);
        v16bf b1 = frag_ld_b(brow1 + kb + 16 * hs);
        acc1 = WMMA_BF16(a, b1, acc1);
    }

    // z = x + h + b_out, staged in LDS for the row-wise LayerNorm
#pragma unroll
    for (int r = 0; r < 8; ++r) {
        int m  = r + 8 * hs;
        int gr = tile * 16 + m;
        zsh[m][n0] = acc0[r] + b_out[n0] + x[(size_t)gr * HID + n0];
        zsh[m][n1] = acc1[r] + b_out[n1] + x[(size_t)gr * HID + n1];
    }
    __syncthreads();

    const int row = threadIdx.x >> 4;    // 16 rows x 16 threads/row
    const int seg = threadIdx.x & 15;
    float s1 = 0.f, s2 = 0.f;
#pragma unroll
    for (int i = 0; i < 16; ++i) {
        float v = zsh[row][seg * 16 + i];
        s1 += v; s2 += v * v;
    }
    red1[row][seg] = s1; red2[row][seg] = s2;
    __syncthreads();
    if (seg == 0) {
        float a = 0.f, b2 = 0.f;
#pragma unroll
        for (int i = 0; i < 16; ++i) { a += red1[row][i]; b2 += red2[row][i]; }
        float mu  = a * (1.0f / HID);
        float var = b2 * (1.0f / HID) - mu * mu;
        red1[row][16] = mu;
        red2[row][16] = __frsqrt_rn(var + 1e-5f);
    }
    __syncthreads();
    const float mu  = red1[row][16];
    const float inv = red2[row][16];
    const int gr = tile * 16 + row;
#pragma unroll
    for (int i = 0; i < 16; ++i) {
        int c = seg * 16 + i;
        out[(size_t)gr * HID + c] = (zsh[row][c] - mu) * inv * ln_g[c] + ln_b[c];
    }
}

// ---------------- launcher ----------------
extern "C" void kernel_launch(void* const* d_in, const int* in_sizes, int n_in,
                              void* d_out, int out_size, void* d_ws, size_t ws_size,
                              hipStream_t stream) {
    const float* x    = (const float*)d_in[0];
    // d_in[1], d_in[2]: stab_rows/cols — analytic (cell index == stabilizer index)
    const float* Win  = (const float*)d_in[3];
    const float* b_in = (const float*)d_in[4];
    const float* w0   = (const float*)d_in[5];
    const float* cb0  = (const float*)d_in[6];
    const float* g0   = (const float*)d_in[7];
    const float* be0  = (const float*)d_in[8];
    const float* w1   = (const float*)d_in[9];
    const float* cb1  = (const float*)d_in[10];
    const float* g1   = (const float*)d_in[11];
    const float* be1  = (const float*)d_in[12];
    const float* Wout = (const float*)d_in[13];
    const float* bo   = (const float*)d_in[14];
    const float* lng  = (const float*)d_in[15];
    const float* lnb  = (const float*)d_in[16];
    float* out = (float*)d_out;

    char* ws = (char*)d_ws;
    size_t off = 0;
    auto alloc = [&](size_t bytes) -> void* {
        void* p = ws + off;
        off = (off + bytes + 255) & ~(size_t)255;
        return p;
    };
    const size_t gridBytesF32 = (size_t)BATCH * GG * CDIM * 4;   // 44.3 MB
    const size_t gridBytesBF  = (size_t)BATCH * GG * CDIM * 2;   // 22.2 MB
    float*  gridA = (float*)alloc(gridBytesF32);
    float*  ybuf  = (float*)alloc(gridBytesF32);
    __bf16* gbf   = (__bf16*)alloc(gridBytesBF);
    __bf16* xbf   = (__bf16*)alloc((size_t)ROWS * HID * 2);      // 40.9 MB
    __bf16* WinT  = (__bf16*)alloc((size_t)CDIM * HID * 2);
    __bf16* WoutT = (__bf16*)alloc((size_t)HID * CDIM * 2);
    __bf16* w0r   = (__bf16*)alloc((size_t)9 * CDIM * CDIM * 2);
    __bf16* w1r   = (__bf16*)alloc((size_t)9 * CDIM * CDIM * 2);
    float*  stats = (float*)alloc(2 * CDIM * 4);

    convert_x_kernel<<<(ROWS * HID) / (256 * 8), 256, 0, stream>>>(x, xbf);
    prep_weights_kernel<<<576, 256, 0, stream>>>(Win, w0, w1, Wout, WinT, w0r, w1r, WoutT);
    hipMemsetAsync(gridA, 0, gridBytesF32, stream);
    hipMemsetAsync(gbf,   0, gridBytesBF,  stream);
    hipMemsetAsync(stats, 0, 2 * CDIM * 4, stream);

    proj_in_kernel<<<ROWS / 64, 256, 0, stream>>>(xbf, WinT, b_in, gridA, gbf);

    conv_kernel<<<BATCH * PT64, 256, 0, stream>>>(gbf, w0r, cb0, ybuf, stats, 1);
    bn_apply_kernel<<<(BATCH * GG * CDIM) / 256, 256, 0, stream>>>(ybuf, stats, g0, be0, gridA, gbf);

    hipMemsetAsync(stats, 0, 2 * CDIM * 4, stream);
    conv_kernel<<<BATCH * PT64, 256, 0, stream>>>(gbf, w1r, cb1, ybuf, stats, 2);
    bn_apply_kernel<<<(BATCH * GG * CDIM) / 256, 256, 0, stream>>>(ybuf, stats, g1, be1, gridA, gbf);

    proj_out_ln_kernel<<<MTILES, 256, 0, stream>>>(x, gbf, WoutT, bo, lng, lnb, out);
}
